// MoEFFN_57698590654857
// MI455X (gfx1250) — matrix operand order repaired and checked by reading, not compiled, not run on hardware
//
#include <hip/hip_runtime.h>
#include <hip/hip_bf16.h>
#include <math.h>

// dims: B=2, T=4096 -> 8192 tokens; D=1024, H=4096, E=8, K=2
#define NTOK 8192
#define DIM  1024
#define HID  4096
#define NEXP 8
#define TM   32      // tokens per tile
#define HB   64      // H chunk
#define KS   128     // phase-1 K slab staged in LDS
#define TILES 256    // ceil(NTOK/TM) worst case (all tokens -> one expert)

// LDS pitches (bf16 elements) -- chosen for 16B alignment of every fragment
// base and distinct bank groups across the 16 fragment rows of a lane group.
#define XS_P  1032   // xs  : 32 x 1032   (row stride 2064B = 516 dw -> +4 banks/row)
#define HS_P  72     // hs  : 32 x 72     (144B = 36 dw)
#define W1_P  136    // w1s : 64 x 136    (272B = 68 dw)
#define W2_P  40     // w2s : 1024 x 40   (80B  = 20 dw)

// dynamic LDS offsets (bytes, all 16B aligned)
#define OFF_XS   0
#define OFF_HS   (OFF_XS + TM * XS_P * 2)          // 66048
#define OFF_W1   (OFF_HS + TM * HS_P * 2)          // 70656
#define OFF_W2   (OFF_W1 + HB * W1_P * 2)          // 88064
#define OFF_TOK  (OFF_W2 + DIM * W2_P * 2)         // 169984
#define OFF_WT   (OFF_TOK + 128)
#define LDS_BYTES (OFF_WT + 128)                   // 170240

typedef __attribute__((ext_vector_type(16))) __bf16 v16bf;
typedef __attribute__((ext_vector_type(8)))  __bf16 v8bf;
typedef __attribute__((ext_vector_type(8)))  float  v8f;

__device__ __forceinline__ __bf16 f2bf(float f) { return (__bf16)f; }

__device__ __forceinline__ v8bf ld8(const __bf16* p) {
    return *(const v8bf*)p;              // 16B aligned -> ds_load_b128
}
__device__ __forceinline__ v16bf cat8(v8bf lo, v8bf hi) {
    return __builtin_shufflevector(lo, hi, 0, 1, 2, 3, 4, 5, 6, 7,
                                           8, 9, 10, 11, 12, 13, 14, 15);
}

// Branchless erf (Abramowitz-Stegun 7.1.26, |err| < 1.5e-7): no exec-mask
// divergence in the hot loop, unlike libm erff.
__device__ __forceinline__ float fast_erf(float x) {
    const float ax = fabsf(x);
    const float t  = 1.0f / (1.0f + 0.3275911f * ax);
    float p = 1.061405429f;
    p = p * t - 1.453152027f;
    p = p * t + 1.421413741f;
    p = p * t - 0.284496736f;
    p = p * t + 0.254829592f;
    const float y = 1.0f - p * t * __expf(-ax * ax);
    return copysignf(y, x);
}
__device__ __forceinline__ float gelu_exact(float v) {
    return 0.5f * v * (1.0f + fast_erf(v * 0.70710678118654752f));
}

// ---------------------------------------------------------------------------
// Kernel 0: zero output (we accumulate with atomics) + zero expert counters
// ---------------------------------------------------------------------------
__global__ __launch_bounds__(256) void moe_zero(float* __restrict__ out, int n,
                                                int* __restrict__ ecnt) {
    int i = blockIdx.x * blockDim.x + threadIdx.x;
    if (i < n) out[i] = 0.0f;
    if (i < NEXP) ecnt[i] = 0;
}

// ---------------------------------------------------------------------------
// Kernel 1: router. One wave (32 lanes) per token. 8 logits over D=1024,
// wave reduction, softmax -> top-2 -> renormalized weights, append token to
// per-expert list with atomic counters.
// ---------------------------------------------------------------------------
__global__ __launch_bounds__(256) void moe_router(const float* __restrict__ x,
                                                  const float* __restrict__ Wr,
                                                  int*   __restrict__ ecnt,
                                                  int*   __restrict__ etok,
                                                  float* __restrict__ ewt) {
    const int wave = threadIdx.x >> 5;
    const int lane = threadIdx.x & 31;
    const int tok  = blockIdx.x * 8 + wave;
    if (tok >= NTOK) return;

    const float* xt = x + (size_t)tok * DIM;
    float acc[NEXP];
#pragma unroll
    for (int e = 0; e < NEXP; ++e) acc[e] = 0.0f;

    for (int d = lane; d < DIM; d += 32) {
        const float xv = xt[d];
#pragma unroll
        for (int e = 0; e < NEXP; ++e) acc[e] += xv * Wr[d * NEXP + e];
    }
#pragma unroll
    for (int off = 16; off > 0; off >>= 1) {
#pragma unroll
        for (int e = 0; e < NEXP; ++e) acc[e] += __shfl_xor(acc[e], off, 32);
    }

    if (lane == 0) {
        float m = acc[0];
#pragma unroll
        for (int e = 1; e < NEXP; ++e) m = fmaxf(m, acc[e]);
        float p[NEXP];
#pragma unroll
        for (int e = 0; e < NEXP; ++e) p[e] = __expf(acc[e] - m);
        int i0 = 0;
#pragma unroll
        for (int e = 1; e < NEXP; ++e) if (p[e] > p[i0]) i0 = e;
        int i1 = (i0 == 0) ? 1 : 0;
#pragma unroll
        for (int e = 0; e < NEXP; ++e) if (e != i0 && p[e] > p[i1]) i1 = e;
        const float ws = p[i0] + p[i1];
        const float w0 = p[i0] / ws;
        const float w1 = p[i1] / ws;
        int s0 = atomicAdd(&ecnt[i0], 1);
        etok[i0 * NTOK + s0] = tok;  ewt[i0 * NTOK + s0] = w0;
        int s1 = atomicAdd(&ecnt[i1], 1);
        etok[i1 * NTOK + s1] = tok;  ewt[i1 * NTOK + s1] = w1;
    }
}

// ---------------------------------------------------------------------------
// Kernel 2: grouped expert FFN, bf16 WMMA fed entirely from LDS.
// grid.x = NEXP * TILES ; block = 256 (8 wave32). Per tile of TM=32 tokens:
//   - gathered x rows staged once to LDS as bf16 (A fragments: 2x ds_load_b128)
//   - W1/W2 k-slabs staged cooperatively (coalesced float4) into transposed
//     (K-contiguous) bf16 LDS layouts (B fragments: 2x ds_load_b128)
//   - phase 1: h = gelu(x W1 + b1) 32x64 tile -> LDS
//   - phase 2: each wave accumulates a 32x128 slice of y = h W2 (16 v8f accs)
//   - epilogue: out[tok] += w_t * (y + b2) via atomicAdd (each token hits 2
//     experts)
// ---------------------------------------------------------------------------
__global__ __launch_bounds__(256) void moe_ffn(const float* __restrict__ x,
                                               const float* __restrict__ W1,
                                               const float* __restrict__ b1,
                                               const float* __restrict__ W2,
                                               const float* __restrict__ b2,
                                               const int*   __restrict__ ecnt,
                                               const int*   __restrict__ etok,
                                               const float* __restrict__ ewt,
                                               float* __restrict__ out) {
    const int e    = blockIdx.x / TILES;
    const int tile = blockIdx.x % TILES;
    const int cnt  = ecnt[e];
    const int row0 = tile * TM;
    if (row0 >= cnt) return;
    const int rows = min(TM, cnt - row0);

    extern __shared__ __align__(16) char smem[];
    __bf16* xs    = (__bf16*)(smem + OFF_XS);
    __bf16* hs    = (__bf16*)(smem + OFF_HS);
    __bf16* w1s   = (__bf16*)(smem + OFF_W1);
    __bf16* w2s   = (__bf16*)(smem + OFF_W2);
    int*    tok_s = (int*)  (smem + OFF_TOK);
    float*  wt_s  = (float*)(smem + OFF_WT);

    if (threadIdx.x < TM) {
        const int g = row0 + (int)threadIdx.x;
        tok_s[threadIdx.x] = (g < cnt) ? etok[e * NTOK + g] : 0;   // pad row 0
        wt_s[threadIdx.x]  = (g < cnt) ? ewt[e * NTOK + g] : 0.0f;
    }
    __syncthreads();

    // ---- stage gathered x tile once: 32 rows x 1024, float4 coalesced ----
    for (int i = threadIdx.x; i < TM * (DIM / 4); i += 256) {
        const int r  = i >> 8;                 // DIM/4 = 256 float4 per row
        const int c4 = i & 255;
        const float4 v =
            ((const float4*)(x + (size_t)tok_s[r] * DIM))[c4];
        __bf16* dst = xs + r * XS_P + c4 * 4;
        dst[0] = f2bf(v.x); dst[1] = f2bf(v.y);
        dst[2] = f2bf(v.z); dst[3] = f2bf(v.w);
    }

    const int wave = threadIdx.x >> 5;
    const int lane = threadIdx.x & 31;
    const int half = lane >> 4;       // lane group (0: lanes 0-15, 1: 16-31)
    const int nl   = lane & 15;       // N / M sub-index within lane group

    const float* w1p = W1 + (size_t)e * DIM * HID;
    const float* w2p = W2 + (size_t)e * HID * DIM;

    // phase-1 role: wave -> (mi, ni) 16x16 subtile of the 32x64 h tile
    const int mi = wave >> 2;         // row tile 0..1
    const int ni = wave & 3;          // col tile 0..3
    // phase-2 role: wave owns 128 output columns
    const int dcol0 = wave * 128;

    v8f yacc[2][8];
#pragma unroll
    for (int t = 0; t < 2; ++t)
#pragma unroll
        for (int c = 0; c < 8; ++c) yacc[t][c] = (v8f){};

    for (int hc = 0; hc < HID / HB; ++hc) {
        // ================ phase 1: h = gelu(x W1 + b1) =====================
        const int hcol = hc * HB + ni * 16 + nl;
        v8f acc = (v8f){};
        for (int ks = 0; ks < DIM; ks += KS) {
            __syncthreads();   // protect w1s from previous consumers
            // stage W1[ks..ks+127][hc*64..+63] transposed -> w1s[h][k]
            for (int i = threadIdx.x; i < KS * (HB / 4); i += 256) {
                const int kl = i >> 4;             // 0..127
                const int h4 = i & 15;             // 16 float4 across 64 h
                const float4 v = *(const float4*)(
                    w1p + (size_t)(ks + kl) * HID + hc * HB + h4 * 4);
                w1s[(h4 * 4 + 0) * W1_P + kl] = f2bf(v.x);
                w1s[(h4 * 4 + 1) * W1_P + kl] = f2bf(v.y);
                w1s[(h4 * 4 + 2) * W1_P + kl] = f2bf(v.z);
                w1s[(h4 * 4 + 3) * W1_P + kl] = f2bf(v.w);
            }
            __syncthreads();
#pragma unroll
            for (int k2 = 0; k2 < KS; k2 += 32) {
                const __bf16* ap = xs + (mi * 16 + nl) * XS_P + ks + k2;
                const v16bf a = cat8(ld8(ap + half * 8),
                                     ld8(ap + 16 + half * 8));
                const __bf16* bp =
                    w1s + (ni * 16 + nl) * W1_P + k2;
                const v16bf b = cat8(ld8(bp + half * 8), ld8(bp + 16 + half * 8));
                acc = __builtin_amdgcn_wmma_f32_16x16x32_bf16(
                          false, a, false, b, (short)0, acc, false, false);
            }
        }
        const float bias = b1[e * HID + hcol];
#pragma unroll
        for (int r = 0; r < 8; ++r) {
            const int m = mi * 16 + r + 8 * half;  // C/D row mapping
            hs[m * HS_P + ni * 16 + nl] = f2bf(gelu_exact(acc[r] + bias));
        }
        __syncthreads();

        // ================ phase 2: y += h W2 over this H chunk =============
#pragma unroll
        for (int kk = 0; kk < HB; kk += 32) {
            // stage W2[hc*64+kk .. +31][0..1023] transposed -> w2s[d][h]
            for (int i = threadIdx.x; i < 32 * (DIM / 4); i += 256) {
                const int hl = i >> 8;             // 0..31
                const int d4 = i & 255;
                const float4 v = *(const float4*)(
                    w2p + (size_t)(hc * HB + kk + hl) * DIM + d4 * 4);
                w2s[(d4 * 4 + 0) * W2_P + hl] = f2bf(v.x);
                w2s[(d4 * 4 + 1) * W2_P + hl] = f2bf(v.y);
                w2s[(d4 * 4 + 2) * W2_P + hl] = f2bf(v.z);
                w2s[(d4 * 4 + 3) * W2_P + hl] = f2bf(v.w);
            }
            __syncthreads();
            const __bf16* h0 = hs + nl * HS_P + kk;
            const __bf16* h1 = hs + (16 + nl) * HS_P + kk;
            const v16bf a0 = cat8(ld8(h0 + half * 8), ld8(h0 + 16 + half * 8));
            const v16bf a1 = cat8(ld8(h1 + half * 8), ld8(h1 + 16 + half * 8));
#pragma unroll
            for (int cj = 0; cj < 8; ++cj) {
                const __bf16* bp =
                    w2s + (dcol0 + cj * 16 + nl) * W2_P;
                const v16bf b = cat8(ld8(bp + half * 8), ld8(bp + 16 + half * 8));
                yacc[0][cj] = __builtin_amdgcn_wmma_f32_16x16x32_bf16(
                                  false, a0, false, b, (short)0, yacc[0][cj],
                                  false, false);
                yacc[1][cj] = __builtin_amdgcn_wmma_f32_16x16x32_bf16(
                                  false, a1, false, b, (short)0, yacc[1][cj],
                                  false, false);
            }
            __syncthreads();   // w2s / hs reused
        }
    }

    // ---------------- epilogue: weighted atomic accumulation ---------------
#pragma unroll
    for (int t = 0; t < 2; ++t) {
#pragma unroll
        for (int cj = 0; cj < 8; ++cj) {
            const int dcol = dcol0 + cj * 16 + nl;
            const float b2v = b2[e * DIM + dcol];
#pragma unroll
            for (int r = 0; r < 8; ++r) {
                const int m = t * 16 + r + 8 * half;
                if (m < rows) {
                    atomicAdd(&out[(size_t)tok_s[m] * DIM + dcol],
                              wt_s[m] * (yacc[t][cj][r] + b2v));
                }
            }
        }
    }
}

// ---------------------------------------------------------------------------
extern "C" void kernel_launch(void* const* d_in, const int* in_sizes, int n_in,
                              void* d_out, int out_size, void* d_ws, size_t ws_size,
                              hipStream_t stream) {
    const float* x  = (const float*)d_in[0];
    const float* Wr = (const float*)d_in[1];
    const float* W1 = (const float*)d_in[2];
    const float* b1 = (const float*)d_in[3];
    const float* W2 = (const float*)d_in[4];
    const float* b2 = (const float*)d_in[5];
    float* out = (float*)d_out;

    // workspace layout: counters | token lists | weight lists  (~0.5 MB)
    int*   ecnt = (int*)d_ws;
    int*   etok = ecnt + 16;
    float* ewt  = (float*)(etok + NEXP * NTOK);

    const int n = NTOK * DIM;
    moe_zero<<<(n + 255) / 256, 256, 0, stream>>>(out, n, ecnt);
    moe_router<<<NTOK / 8, 256, 0, stream>>>(x, Wr, ecnt, etok, ewt);
    moe_ffn<<<NEXP * TILES, 256, LDS_BYTES, stream>>>(x, W1, b1, W2, b2,
                                                      ecnt, etok, ewt, out);
}
